// CustomDynamicEdgeConv_996432413187
// MI455X (gfx1250) — matrix operand order repaired and checked by reading, not compiled, not run on hardware
//
#include <hip/hip_runtime.h>
#include <hip/hip_bf16.h>
#include <stdint.h>

// Problem sizes (static, match reference)
#define BGR  64
#define NPG  1024
#define NTOT (BGR * NPG)   // 65536
#define FDIM 64
#define HDIM 128
#define KNN  16

typedef __attribute__((ext_vector_type(16))) __bf16 bf16x16;
typedef __attribute__((ext_vector_type(8)))  float  f32x8;
typedef __attribute__((ext_vector_type(4)))  unsigned v4u;
typedef __attribute__((ext_vector_type(8)))  int      v8i;
typedef __attribute__((ext_vector_type(4)))  int      v4i;

// round-to-nearest-even f32 -> bf16 bits (pure integer ops)
__device__ __forceinline__ unsigned short f2bf(float f) {
  unsigned u = __float_as_uint(f);
  u += 0x7FFFu + ((u >> 16) & 1u);
  return (unsigned short)(u >> 16);
}
__device__ __forceinline__ float bf2f(unsigned short b) {
  return __uint_as_float((unsigned)b << 16);
}

// generic (flat) shared pointer -> 32-bit LDS byte offset for TDM D#
typedef __attribute__((address_space(3))) void lds_void;
__device__ __forceinline__ unsigned lds_off(void* p) {
  return (unsigned)(unsigned long long)(lds_void*)p;
}

// ---------------------------------------------------------------- TDM
// Issue a 2D TENSOR_LOAD_TO_LDS: d1 rows of d0 bf16 elements, contiguous in
// global (row stride = d0), optionally padding the LDS destination after each
// (1<<pad_int_code)*2 dwords by (pad_amt_code+1) dwords.
// D# layout per CDNA5 ISA. Wave-uniform args only.
// This toolchain exposes the 6-arg builtin (g0, g1, g2, g3, g4, cpol).
__device__ __forceinline__ void tdm_load_2d(unsigned lds_addr, const void* gptr,
                                            unsigned d0, unsigned d1,
                                            unsigned pad_int_code, unsigned pad_amt_code,
                                            bool pad_en) {
  unsigned long long ga = (unsigned long long)gptr;
  v4u g0;
  g0[0] = 1u;                                            // count=1, user mode
  g0[1] = lds_addr;                                      // lds_addr
  g0[2] = (unsigned)(ga & 0xFFFFFFFFu);                  // global_addr lo
  g0[3] = (unsigned)((ga >> 32) & 0x01FFFFFFu) | (2u << 30);  // addr hi | type=2
  unsigned w0 = (1u << 16);                              // data_size = 1 -> 2 bytes
  if (pad_en) w0 |= (1u << 20) | (pad_int_code << 22) | (pad_amt_code << 25);
  v8i g1;
  g1[0] = (int)w0;                                       // mask=0, flags
  g1[1] = (int)((d0 & 0xFFFFu) << 16);                   // tensor_dim0 lo16
  g1[2] = (int)(((d0 >> 16) & 0xFFFFu) | ((d1 & 0xFFFFu) << 16)); // dim0 hi | dim1 lo
  g1[3] = (int)(((d1 >> 16) & 0xFFFFu) | ((d0 & 0xFFFFu) << 16)); // dim1 hi | tile_dim0
  g1[4] = (int)(d1 & 0xFFFFu);                           // tile_dim1 | tile_dim2=0
  g1[5] = (int)d0;                                       // tensor_dim0_stride lo32
  g1[6] = 0;                                             // stride hi | dim1_stride lo
  g1[7] = 0;
  v4i z4 = {0, 0, 0, 0};
  v8i z8 = {0, 0, 0, 0, 0, 0, 0, 0};
  __builtin_amdgcn_tensor_load_to_lds(g0, g1, z4, z4, z8, 0);
}

// A-matrix 16x32 bf16 fragment (ISA 7.12.2): lanes 0-15 hold K=0..7,16..23;
// lanes 16-31 hold K=8..15,24..31 for row M = lane&15. Source: [row][k] LDS.
__device__ __forceinline__ bf16x16 load_fragA(const unsigned short* base, int r0, int k0, int stride) {
  const int lane = threadIdx.x & 31;
  const int half = lane >> 4;
  const int m    = lane & 15;
  const unsigned short* p = base + (r0 + m) * stride + k0 + half * 8;
  union { bf16x16 v; unsigned u[8]; } f;
#pragma unroll
  for (int i = 0; i < 8; ++i) {
    const int kk = (i < 4) ? (2 * i) : (16 + 2 * (i - 4));
    f.u[i] = *(const unsigned*)(p + kk);
  }
  return f.v;
}

// B-matrix 32x16 bf16 fragment (same K order as A): lanes 0-15
// hold K=0..7,16..23, lanes 16-31 hold K=8..15,24..31 for column N = lane&15.
// Source: [n][k] LDS (i.e. W^T row-major).
__device__ __forceinline__ bf16x16 load_fragB(const unsigned short* base, int n0, int k0, int stride) {
  const int lane = threadIdx.x & 31;
  const int half = lane >> 4;
  const int n    = lane & 15;
  const unsigned short* p = base + (n0 + n) * stride + k0 + half * 8;
  union { bf16x16 v; unsigned u[8]; } f;
#pragma unroll
  for (int i = 0; i < 8; ++i) f.u[i] = *(const unsigned*)(p + ((i < 4) ? (2 * i) : (16 + 2 * (i - 4))));
  return f.v;
}

// ---------------------------------------------------------------- utilities
__global__ void zero_kernel(float* __restrict__ out, unsigned* __restrict__ cnt) {
  long long i = (long long)blockIdx.x * blockDim.x + threadIdx.x;
  if (i < (long long)NTOT * HDIM) out[i] = 0.0f;
  if (i < NTOT) cnt[i] = 0u;
}

// one-shot f32 -> bf16 copy of x (halves all later staging traffic)
__global__ void xbf_kernel(const float* __restrict__ x, unsigned short* __restrict__ xbf) {
  int i = blockIdx.x * blockDim.x + threadIdx.x;
  xbf[i] = f2bf(x[i]);
}

__global__ void norms_kernel(const float* __restrict__ x, float* __restrict__ sqn) {
  int i = blockIdx.x * blockDim.x + threadIdx.x;
  if (i >= NTOT) return;
  const float4* p = (const float4*)(x + (long long)i * FDIM);
  float s = 0.0f;
#pragma unroll
  for (int j = 0; j < FDIM / 4; ++j) {
    float4 v = p[j];
    s += v.x * v.x + v.y * v.y + v.z * v.z + v.w * v.w;
  }
  sqn[i] = s;
}

// Pack W1,W2 (fp32 [K][N]) into bf16 W^T ([N][K]) so B-fragment K-pairs are
// contiguous dwords in LDS.
__global__ void prep_weights_kernel(const float* __restrict__ W1, const float* __restrict__ W2,
                                    unsigned short* __restrict__ w1t, unsigned short* __restrict__ w2t) {
  int idx = blockIdx.x * blockDim.x + threadIdx.x;
  if (idx >= HDIM * HDIM) return;
  int n = idx >> 7, k = idx & 127;
  w1t[n * HDIM + k] = f2bf(W1[k * HDIM + n]);
  w2t[n * HDIM + k] = f2bf(W2[k * HDIM + n]);
}

// ---------------------------------------------------------------- KNN
// Block: 256 threads (8 waves), handles 128 rows of one graph. TDM stages the
// bf16 feature tiles (pad 8 elems per 64-elem row -> LDS stride 72); WMMA
// computes 128x128 dot tiles; 2 lanes/row keep a register top-16.
__global__ __launch_bounds__(256) void knn_kernel(const unsigned short* __restrict__ xbf,
                                                  const float* __restrict__ sqn,
                                                  int* __restrict__ nbr,
                                                  unsigned* __restrict__ cnt) {
  const int g    = blockIdx.y;
  const int rt   = blockIdx.x;
  const int t    = threadIdx.x;
  const int wave = t >> 5;

  __shared__ unsigned short rx[128][FDIM + 8];   // row features, bf16, padded
  __shared__ unsigned short cx[128][FDIM + 8];   // col-chunk features
  __shared__ float dist[128][128 + 4];           // dot products (pad vs banks)
  __shared__ float sqc[128];
  __shared__ float md[128][KNN];                 // merge buffers
  __shared__ int   mi[128][KNN];

  const int       rowBase = rt * 128;
  const long long gbase   = (long long)g * NPG;

  // stage 128x64 bf16 row tile via TDM: pad 4 dwords after every 16x8B=128B row
  if (t < 32) {
    tdm_load_2d(lds_off(&rx[0][0]), xbf + (gbase + rowBase) * FDIM,
                FDIM, 128, /*interval 128B*/4, /*amount 4dw*/3, true);
    __builtin_amdgcn_s_wait_tensorcnt(0);
  }
  __syncthreads();

  // per-wave A fragments for its M-tile are invariant over the column loop
  bf16x16 a0 = load_fragA(&rx[0][0], wave * 16, 0,  FDIM + 8);
  bf16x16 a1 = load_fragA(&rx[0][0], wave * 16, 32, FDIM + 8);

  const int   r        = t & 127;
  const int   part     = t >> 7;      // split scan: 2 lanes per row
  const int   rowLocal = rowBase + r;
  const float sqr      = sqn[gbase + rowLocal];

  float dk[KNN]; int ik[KNN];
#pragma unroll
  for (int j = 0; j < KNN; ++j) { dk[j] = 3.0e38f; ik[j] = 0; }

  for (int ct = 0; ct < NPG / 128; ++ct) {
    const int colbase = ct * 128;
    __syncthreads();                               // prev dist consumed, cx free
    if (t < 32) {
      tdm_load_2d(lds_off(&cx[0][0]), xbf + (gbase + colbase) * FDIM,
                  FDIM, 128, 4, 3, true);
      __builtin_amdgcn_s_wait_tensorcnt(0);
    }
    if (t < 128) sqc[t] = sqn[gbase + colbase + t];
    __syncthreads();

#pragma unroll
    for (int n0 = 0; n0 < 8; ++n0) {
      bf16x16 b0 = load_fragB(&cx[0][0], n0 * 16, 0,  FDIM + 8);
      bf16x16 b1 = load_fragB(&cx[0][0], n0 * 16, 32, FDIM + 8);
      f32x8 acc = {0.f, 0.f, 0.f, 0.f, 0.f, 0.f, 0.f, 0.f};
      acc = __builtin_amdgcn_wmma_f32_16x16x32_bf16(false, a0, false, b0, (short)0, acc, false, false);
      acc = __builtin_amdgcn_wmma_f32_16x16x32_bf16(false, a1, false, b1, (short)0, acc, false, false);
      const int lane = t & 31;
      const int col  = n0 * 16 + (lane & 15);
      const int half = lane >> 4;
#pragma unroll
      for (int v = 0; v < 8; ++v)
        dist[wave * 16 + v + 8 * half][col] = acc[v];
    }
    __syncthreads();

    for (int cc = 0; cc < 64; ++cc) {
      const int c  = part * 64 + cc;
      const int gc = colbase + c;
      if (gc == rowLocal) continue;                       // no self loops
      const float d = sqr + sqc[c] - 2.0f * dist[r][c];
      if (d < dk[KNN - 1]) {
        float nd = d; int ni = gc;
#pragma unroll
        for (int j = 0; j < KNN; ++j) {
          const bool lt = nd < dk[j];
          const float td = dk[j]; const int ti = ik[j];
          dk[j] = lt ? nd : td; ik[j] = lt ? ni : ti;
          nd = lt ? td : nd;    ni = lt ? ti : ni;
        }
      }
    }
  }

  // merge the two half-scans per row
  if (t >= 128) {
#pragma unroll
    for (int j = 0; j < KNN; ++j) { md[r][j] = dk[j]; mi[r][j] = ik[j]; }
  }
  __syncthreads();
  if (t < 128) {
#pragma unroll
    for (int j = 0; j < KNN; ++j) {
      float nd = md[r][j]; int ni = mi[r][j];
      if (nd < dk[KNN - 1]) {
#pragma unroll
        for (int q = 0; q < KNN; ++q) {
          const bool lt = nd < dk[q];
          const float td = dk[q]; const int ti = ik[q];
          dk[q] = lt ? nd : td; ik[q] = lt ? ni : ti;
          nd = lt ? td : nd;    ni = lt ? ti : ni;
        }
      }
    }
    const long long center = gbase + rowLocal;
#pragma unroll
    for (int j = 0; j < KNN; ++j) {
      const int nid = (int)gbase + ik[j];
      nbr[center * KNN + j] = nid;
      atomicAdd(&cnt[nid], 1u);   // row (=neighbor) in-degree for scatter-mean
    }
  }
}

// ---------------------------------------------------------------- fused MLP
// Block: 256 threads (8 waves) = 8 centers = 128 edges; each wave owns one
// 16-edge M-tile through both layers. Weights arrive via TDM (pad 4 dwords
// per 256B row -> LDS stride 136); activations stay resident in LDS.
__global__ __launch_bounds__(256) void edge_mlp_kernel(const unsigned short* __restrict__ xbf,
                                                       const unsigned short* __restrict__ w1t,
                                                       const unsigned short* __restrict__ w2t,
                                                       const float* __restrict__ b1,
                                                       const float* __restrict__ b2,
                                                       const int* __restrict__ nbr,
                                                       float* __restrict__ out) {
  const int t    = threadIdx.x;
  const int wave = t >> 5;
  const int c0   = blockIdx.x * 8;

  __shared__ unsigned short sW1[HDIM][HDIM + 8];  // W1^T [n][k]
  __shared__ unsigned short sW2[HDIM][HDIM + 8];  // W2^T [n][k]
  __shared__ unsigned short sef[128][HDIM + 8];   // edge features [e][k]
  __shared__ unsigned short sh1[128][HDIM + 8];   // layer-1 activations
  __shared__ float sb1[HDIM], sb2[HDIM];
  __shared__ int sn[128];

  if (t < 32) {
    tdm_load_2d(lds_off(&sW1[0][0]), w1t, HDIM, HDIM, /*interval 256B*/5, /*amount 4dw*/3, true);
    tdm_load_2d(lds_off(&sW2[0][0]), w2t, HDIM, HDIM, 5, 3, true);
    __builtin_amdgcn_s_wait_tensorcnt(0);
  }
  if (t < HDIM) { sb1[t] = b1[t]; sb2[t] = b2[t]; }
  if (t < 128) sn[t] = nbr[(long long)(c0 + (t >> 4)) * KNN + (t & 15)];
  __syncthreads();

  // ef = [x[n], x[c] - x[n]]   (row = neighbor per reference)
  for (int idx = t; idx < 128 * HDIM; idx += 256) {
    const int e = idx >> 7, k = idx & 127;
    const int c = c0 + (e >> 4);
    const int n = sn[e];
    if (k < FDIM) {
      sef[e][k] = xbf[n * FDIM + k];                          // raw bf16 copy
    } else {
      const int kk = k - FDIM;
      sef[e][k] = f2bf(bf2f(xbf[c * FDIM + kk]) - bf2f(xbf[n * FDIM + kk]));
    }
  }
  __syncthreads();

  const int lane   = t & 31;
  const int half   = lane >> 4;
  const int colsub = lane & 15;

  // ---- layer 1: relu(ef @ W1 + b1)
  bf16x16 A0 = load_fragA(&sef[0][0], wave * 16, 0,  HDIM + 8);
  bf16x16 A1 = load_fragA(&sef[0][0], wave * 16, 32, HDIM + 8);
  bf16x16 A2 = load_fragA(&sef[0][0], wave * 16, 64, HDIM + 8);
  bf16x16 A3 = load_fragA(&sef[0][0], wave * 16, 96, HDIM + 8);
#pragma unroll
  for (int n0 = 0; n0 < 8; ++n0) {
    const float bv = sb1[n0 * 16 + colsub];
    f32x8 acc = {bv, bv, bv, bv, bv, bv, bv, bv};
    bf16x16 B0 = load_fragB(&sW1[0][0], n0 * 16, 0,  HDIM + 8);
    bf16x16 B1 = load_fragB(&sW1[0][0], n0 * 16, 32, HDIM + 8);
    bf16x16 B2 = load_fragB(&sW1[0][0], n0 * 16, 64, HDIM + 8);
    bf16x16 B3 = load_fragB(&sW1[0][0], n0 * 16, 96, HDIM + 8);
    acc = __builtin_amdgcn_wmma_f32_16x16x32_bf16(false, A0, false, B0, (short)0, acc, false, false);
    acc = __builtin_amdgcn_wmma_f32_16x16x32_bf16(false, A1, false, B1, (short)0, acc, false, false);
    acc = __builtin_amdgcn_wmma_f32_16x16x32_bf16(false, A2, false, B2, (short)0, acc, false, false);
    acc = __builtin_amdgcn_wmma_f32_16x16x32_bf16(false, A3, false, B3, (short)0, acc, false, false);
#pragma unroll
    for (int v = 0; v < 8; ++v) {
      const float h = acc[v] > 0.0f ? acc[v] : 0.0f;
      sh1[wave * 16 + v + 8 * half][n0 * 16 + colsub] = f2bf(h);
    }
  }
  // each wave reads back only the sh1 rows it wrote (same-wave LDS is ordered)

  // ---- layer 2: relu(h1 @ W2 + b2), scatter-add onto neighbor rows
  bf16x16 H0 = load_fragA(&sh1[0][0], wave * 16, 0,  HDIM + 8);
  bf16x16 H1 = load_fragA(&sh1[0][0], wave * 16, 32, HDIM + 8);
  bf16x16 H2 = load_fragA(&sh1[0][0], wave * 16, 64, HDIM + 8);
  bf16x16 H3 = load_fragA(&sh1[0][0], wave * 16, 96, HDIM + 8);
#pragma unroll
  for (int n0 = 0; n0 < 8; ++n0) {
    const float bv = sb2[n0 * 16 + colsub];
    f32x8 acc = {bv, bv, bv, bv, bv, bv, bv, bv};
    bf16x16 B0 = load_fragB(&sW2[0][0], n0 * 16, 0,  HDIM + 8);
    bf16x16 B1 = load_fragB(&sW2[0][0], n0 * 16, 32, HDIM + 8);
    bf16x16 B2 = load_fragB(&sW2[0][0], n0 * 16, 64, HDIM + 8);
    bf16x16 B3 = load_fragB(&sW2[0][0], n0 * 16, 96, HDIM + 8);
    acc = __builtin_amdgcn_wmma_f32_16x16x32_bf16(false, H0, false, B0, (short)0, acc, false, false);
    acc = __builtin_amdgcn_wmma_f32_16x16x32_bf16(false, H1, false, B1, (short)0, acc, false, false);
    acc = __builtin_amdgcn_wmma_f32_16x16x32_bf16(false, H2, false, B2, (short)0, acc, false, false);
    acc = __builtin_amdgcn_wmma_f32_16x16x32_bf16(false, H3, false, B3, (short)0, acc, false, false);
#pragma unroll
    for (int v = 0; v < 8; ++v) {
      const float h = acc[v] > 0.0f ? acc[v] : 0.0f;
      const int e  = wave * 16 + v + 8 * half;
      const int nn = sn[e];
      atomicAdd(&out[(long long)nn * HDIM + n0 * 16 + colsub], h);
    }
  }
}

__global__ void finalize_kernel(float* __restrict__ out, const unsigned* __restrict__ cnt) {
  long long i = (long long)blockIdx.x * blockDim.x + threadIdx.x;
  if (i >= (long long)NTOT * HDIM) return;
  const unsigned c = cnt[i >> 7];
  out[i] *= 1.0f / (float)(c ? c : 1u);
}

// ---------------------------------------------------------------- launch
extern "C" void kernel_launch(void* const* d_in, const int* in_sizes, int n_in,
                              void* d_out, int out_size, void* d_ws, size_t ws_size,
                              hipStream_t stream) {
  (void)in_sizes; (void)n_in; (void)out_size; (void)ws_size;
  const float* x  = (const float*)d_in[0];
  // d_in[1] = batch ids: unused (contiguous NPG blocks per graph by construction)
  const float* W1 = (const float*)d_in[2];
  const float* b1 = (const float*)d_in[3];
  const float* W2 = (const float*)d_in[4];
  const float* b2 = (const float*)d_in[5];
  float* out = (float*)d_out;

  char* ws = (char*)d_ws;
  int*            nbr = (int*)ws;            ws += (size_t)NTOT * KNN * sizeof(int);
  unsigned*       cnt = (unsigned*)ws;       ws += (size_t)NTOT * sizeof(unsigned);
  float*          sqn = (float*)ws;          ws += (size_t)NTOT * sizeof(float);
  unsigned short* w1t = (unsigned short*)ws; ws += (size_t)HDIM * HDIM * sizeof(unsigned short);
  unsigned short* w2t = (unsigned short*)ws; ws += (size_t)HDIM * HDIM * sizeof(unsigned short);
  unsigned short* xbf = (unsigned short*)ws;

  const long long total = (long long)NTOT * HDIM;
  zero_kernel<<<dim3((unsigned)((total + 255) / 256)), 256, 0, stream>>>(out, cnt);
  xbf_kernel<<<dim3(NTOT * FDIM / 256), 256, 0, stream>>>(x, xbf);
  norms_kernel<<<dim3(NTOT / 256), 256, 0, stream>>>(x, sqn);
  prep_weights_kernel<<<dim3(HDIM * HDIM / 256), 256, 0, stream>>>(W1, W2, w1t, w2t);
  knn_kernel<<<dim3(NPG / 128, BGR), 256, 0, stream>>>(xbf, sqn, nbr, cnt);
  edge_mlp_kernel<<<dim3(NTOT / 8), 256, 0, stream>>>(xbf, w1t, w2t, b1, b2, nbr, out);
  finalize_kernel<<<dim3((unsigned)((total + 255) / 256)), 256, 0, stream>>>(out, cnt);
}
